// MultiHeadAttentionLayer_38388417691980
// MI455X (gfx1250) — matrix operand order repaired and checked by reading, not compiled, not run on hardware
//
#include <hip/hip_runtime.h>
#include <hip/hip_bf16.h>
#include <math.h>

// ---------------------------------------------------------------------------
// T5 Multi-Head Attention for MI455X (gfx1250), wave32 + WMMA bf16.
// B=2, S=2048, D_MODEL=1024, H=16, HEAD=64, NUM_BUCKETS=32, MAX_DIST=128.
// ---------------------------------------------------------------------------

typedef __attribute__((ext_vector_type(16))) __bf16 v16bf;
typedef __attribute__((ext_vector_type(8)))  float  v8f;
typedef __attribute__((ext_vector_type(4)))  unsigned int v4u;
typedef __attribute__((ext_vector_type(8)))  unsigned int v8u;

#define DM   1024
#define SEQ  2048
#define NH   16
#define HD   64
#define NB_  2

// ---- fragment load helpers -------------------------------------------------
__device__ __forceinline__ v16bf frag16(const __bf16* p0, const __bf16* p1) {
    v4u a = *reinterpret_cast<const v4u*>(p0);   // 8 bf16
    v4u b = *reinterpret_cast<const v4u*>(p1);   // 8 bf16
    v8u u;
    u[0]=a[0]; u[1]=a[1]; u[2]=a[2]; u[3]=a[3];
    u[4]=b[0]; u[5]=b[1]; u[6]=b[2]; u[7]=b[3];
    return __builtin_bit_cast(v16bf, u);
}
// contiguous 16 bf16 (B-fragment from a row-major B^T row)
__device__ __forceinline__ v16bf fragB(const __bf16* p) { return frag16(p, p + 8); }

__device__ __forceinline__ float hmax16(float x) {
    x = fmaxf(x, __shfl_xor(x, 8, 32));
    x = fmaxf(x, __shfl_xor(x, 4, 32));
    x = fmaxf(x, __shfl_xor(x, 2, 32));
    x = fmaxf(x, __shfl_xor(x, 1, 32));
    return x;
}
__device__ __forceinline__ float hsum16(float x) {
    x += __shfl_xor(x, 8, 32);
    x += __shfl_xor(x, 4, 32);
    x += __shfl_xor(x, 2, 32);
    x += __shfl_xor(x, 1, 32);
    return x;
}

#define WMMA_BF16(A, Bm, C) \
    __builtin_amdgcn_wmma_f32_16x16x32_bf16(false, (A), false, (Bm), (short)0, (C), false, false)

// ---------------------------------------------------------------------------
// Kernel 1: transpose + convert the three projection weights to bf16.
// ---------------------------------------------------------------------------
__global__ void t5_prep_weights(const float* __restrict__ Wq,
                                const float* __restrict__ Wk,
                                const float* __restrict__ Wv,
                                __bf16* __restrict__ wt) {
    int idx = blockIdx.x * blockDim.x + threadIdx.x;       // 3*1024*1024 threads
    int wsel = idx >> 20;
    int r    = idx & (DM * DM - 1);
    int k = r >> 10, c = r & (DM - 1);
    const float* W = (wsel == 0) ? Wq : (wsel == 1) ? Wk : Wv;
    wt[(size_t)wsel * DM * DM + (size_t)c * DM + k] = (__bf16)W[r];
}

// ---------------------------------------------------------------------------
// Kernel 2: convert query/key/value activations to bf16 (row-major, once).
// ---------------------------------------------------------------------------
__global__ void t5_prep_x(const float* __restrict__ xq,
                          const float* __restrict__ xk,
                          const float* __restrict__ xv,
                          __bf16* __restrict__ xbf) {
    size_t idx = (size_t)blockIdx.x * blockDim.x + threadIdx.x; // 3*4096*1024
    int    wsel = (int)(idx >> 22);
    size_t r    = idx & ((1u << 22) - 1);
    const float* X = (wsel == 0) ? xq : (wsel == 1) ? xk : xv;
    xbf[idx] = (__bf16)X[r];
}

// ---------------------------------------------------------------------------
// Kernel 3: relative-position bias table, biasTab[h][d+2047], d = key - query.
// ---------------------------------------------------------------------------
__global__ void t5_prep_bias(const float* __restrict__ bias_table,
                             float* __restrict__ biasTab) {
    int idx = blockIdx.x * blockDim.x + threadIdx.x;
    if (idx >= NH * 4095) return;
    int h = idx / 4095;
    int d = (idx % 4095) - 2047;
    int ret = (d > 0) ? 16 : 0;            // bidirectional, num_buckets/2 = 16
    int rp  = (d < 0) ? -d : d;
    int bucket;
    if (rp < 8) {
        bucket = ret + rp;                  // max_exact = 8
    } else {
        float v = __logf((float)rp * 0.125f) * (8.0f / __logf(16.0f));
        int vl = 8 + (int)v;
        bucket = ret + (vl < 15 ? vl : 15);
    }
    biasTab[idx] = bias_table[bucket * NH + h];
}

// ---------------------------------------------------------------------------
// Kernel 4: QKV projection GEMM. One wave = one 16x64 output strip (one head),
// 4 accumulators reuse each A-fragment across 4 WMMAs.
// Q -> qws[b,h,s,dd]   K -> kws[b,h,s,dd]   V -> vtws[b,h,dd,s] (transposed)
// ---------------------------------------------------------------------------
__global__ void __launch_bounds__(128) t5_proj_gemm(
        const __bf16* __restrict__ xbf, const __bf16* __restrict__ wt,
        __bf16* __restrict__ qws, __bf16* __restrict__ kws,
        __bf16* __restrict__ vtws) {
    int wave = blockIdx.x * 4 + (threadIdx.x >> 5);   // 3*256*16 waves
    int lane = threadIdx.x & 31;
    int wsel = wave >> 12;          // / (256*16)
    int r    = wave & 4095;
    int rt = r >> 4, ctg = r & 15;  // row-tile, head (64-col group)
    int lm = lane & 15, lg = lane >> 4;

    const __bf16* xrow = xbf + ((size_t)wsel << 22) + (size_t)(rt * 16 + lm) * DM;
    const __bf16* wbase = wt + (size_t)wsel * DM * DM
                             + (size_t)(ctg * 64 + lm) * DM;   // + j*16*DM per tile

    v8f acc0 = {}, acc1 = {}, acc2 = {}, acc3 = {};
#pragma unroll 2
    for (int ks = 0; ks < DM / 32; ++ks) {
        int d0 = ks * 32;
        v16bf a = frag16(xrow + d0 + lg * 16, xrow + d0 + lg * 16 + 8);
        acc0 = WMMA_BF16(a, fragB(wbase + 0 * 16 * DM + d0 + lg * 16), acc0);
        acc1 = WMMA_BF16(a, fragB(wbase + 1 * 16 * DM + d0 + lg * 16), acc1);
        acc2 = WMMA_BF16(a, fragB(wbase + 2 * 16 * DM + d0 + lg * 16), acc2);
        acc3 = WMMA_BF16(a, fragB(wbase + 3 * 16 * DM + d0 + lg * 16), acc3);
    }

    int h = ctg;
#pragma unroll
    for (int v = 0; v < 8; ++v) {
        int s  = rt * 16 + v + 8 * lg;
        int bb = s >> 11, sq = s & (SEQ - 1);
        size_t bh = (size_t)(bb * NH + h);
        float av[4] = { acc0[v], acc1[v], acc2[v], acc3[v] };
#pragma unroll
        for (int j = 0; j < 4; ++j) {
            int dd = j * 16 + lm;
            __bf16 val = (__bf16)av[j];
            if (wsel == 0)      qws[(bh * SEQ + sq) * HD + dd] = val;
            else if (wsel == 1) kws[(bh * SEQ + sq) * HD + dd] = val;
            else                vtws[(bh * HD + dd) * SEQ + sq] = val;
        }
    }
}

// ---------------------------------------------------------------------------
// Kernel 5: fused flash attention. One wave = one (b,h, 32-row q-tile) as two
// 16-row sub-tiles A/B; 32-key blocks: 8 score WMMA + 8 context WMMA per
// iteration, K/V fragments each reused by both sub-tiles.
// ---------------------------------------------------------------------------
__global__ void __launch_bounds__(128) t5_flash_attn(
        const __bf16* __restrict__ qws, const __bf16* __restrict__ kws,
        const __bf16* __restrict__ vtws, const float* __restrict__ key_mask,
        const float* __restrict__ biasTab, float* __restrict__ out) {
    __shared__ __align__(64) __bf16 plds[4][2 * 16 * 32];   // 2 P tiles per wave

    int wave = blockIdx.x * 4 + (threadIdx.x >> 5);   // 2*16*64 waves
    int lane = threadIdx.x & 31;
    int b  = wave >> 10;
    int r  = wave & 1023;
    int h  = r >> 6;
    int q0 = (r & 63) * 32;
    int lm = lane & 15, lg = lane >> 4;
    size_t bh = (size_t)(b * NH + h);

    // Q A-fragments for both sub-tiles (live for the whole K loop)
    const __bf16* qrowA = qws + (bh * SEQ + q0 + lm) * HD;
    const __bf16* qrowB = qrowA + 16 * HD;
    v16bf aqA0 = frag16(qrowA + lg * 16,      qrowA + lg * 16 + 8);
    v16bf aqA1 = frag16(qrowA + 32 + lg * 16, qrowA + 40 + lg * 16);
    v16bf aqB0 = frag16(qrowB + lg * 16,      qrowB + lg * 16 + 8);
    v16bf aqB1 = frag16(qrowB + 32 + lg * 16, qrowB + 40 + lg * 16);

    float mA[8], lA[8], mB[8], lB[8];
    v8f OA0 = {}, OA1 = {}, OA2 = {}, OA3 = {};
    v8f OB0 = {}, OB1 = {}, OB2 = {}, OB3 = {};
#pragma unroll
    for (int v = 0; v < 8; ++v) {
        mA[v] = -INFINITY; lA[v] = 0.0f;
        mB[v] = -INFINITY; lB[v] = 0.0f;
    }

    __bf16* pl = plds[threadIdx.x >> 5];
    const float* btbA = biasTab + h * 4095 + 2047 - q0 - 8 * lg;
    const float* btbB = btbA - 16;
    const __bf16* kbase = kws  + (bh * SEQ) * HD;
    const __bf16* vbase = vtws + (bh * HD) * SEQ;

    for (int kt = 0; kt < SEQ / 32; ++kt) {
        int k0 = kt * 32;
        int kp = (k0 + 32) & (SEQ - 1);     // next-tile prefetch (wraps, harmless)
        __builtin_prefetch(kbase + (size_t)(kp + lane) * HD, 0, 1);
        __builtin_prefetch(vbase + (size_t)(lane & (HD - 1)) * SEQ + kp, 0, 1);

        // ---- scores: S[32q x 32k] = Q @ K^T, K frags reused by A/B ------
        const __bf16* krow0 = kbase + (size_t)(k0 + lm) * HD;
        const __bf16* krow1 = krow0 + 16 * HD;
        v8f sA0 = {}, sA1 = {}, sB0 = {}, sB1 = {};
        {
            v16bf kf = fragB(krow0 + lg * 16);
            sA0 = WMMA_BF16(aqA0, kf, sA0);  sB0 = WMMA_BF16(aqB0, kf, sB0);
        }
        {
            v16bf kf = fragB(krow0 + 32 + lg * 16);
            sA0 = WMMA_BF16(aqA1, kf, sA0);  sB0 = WMMA_BF16(aqB1, kf, sB0);
        }
        {
            v16bf kf = fragB(krow1 + lg * 16);
            sA1 = WMMA_BF16(aqA0, kf, sA1);  sB1 = WMMA_BF16(aqB0, kf, sB1);
        }
        {
            v16bf kf = fragB(krow1 + 32 + lg * 16);
            sA1 = WMMA_BF16(aqA1, kf, sA1);  sB1 = WMMA_BF16(aqB1, kf, sB1);
        }

        // ---- + relative bias, + key mask --------------------------------
        {
            int n0 = k0 + lm, n1 = n0 + 16;
            float madd0 = (1.0f - key_mask[b * SEQ + n0]) * -10000.0f;
            float madd1 = (1.0f - key_mask[b * SEQ + n1]) * -10000.0f;
            const float* bpA0 = btbA + n0;  const float* bpA1 = btbA + n1;
            const float* bpB0 = btbB + n0;  const float* bpB1 = btbB + n1;
#pragma unroll
            for (int v = 0; v < 8; ++v) {
                sA0[v] += bpA0[-v] + madd0;  sA1[v] += bpA1[-v] + madd1;
                sB0[v] += bpB0[-v] + madd0;  sB1[v] += bpB1[-v] + madd1;
            }
        }

        // ---- online softmax (per 16-lane row group) ---------------------
#pragma unroll
        for (int v = 0; v < 8; ++v) {
            float xa  = hmax16(fmaxf(sA0[v], sA1[v]));
            float mna = fmaxf(mA[v], xa);
            float sca = __expf(mA[v] - mna);
            mA[v] = mna;
            float pa0 = __expf(sA0[v] - mna), pa1 = __expf(sA1[v] - mna);
            sA0[v] = pa0; sA1[v] = pa1;
            lA[v] = lA[v] * sca + hsum16(pa0 + pa1);
            OA0[v] *= sca; OA1[v] *= sca; OA2[v] *= sca; OA3[v] *= sca;

            float xb  = hmax16(fmaxf(sB0[v], sB1[v]));
            float mnb = fmaxf(mB[v], xb);
            float scb = __expf(mB[v] - mnb);
            mB[v] = mnb;
            float pb0 = __expf(sB0[v] - mnb), pb1 = __expf(sB1[v] - mnb);
            sB0[v] = pb0; sB1[v] = pb1;
            lB[v] = lB[v] * scb + hsum16(pb0 + pb1);
            OB0[v] *= scb; OB1[v] *= scb; OB2[v] *= scb; OB3[v] *= scb;
        }

        // ---- C-layout P -> LDS -> A-layout fragments --------------------
#pragma unroll
        for (int v = 0; v < 8; ++v) {
            int rr = (v + 8 * lg) * 32;
            pl[rr + lm]            = (__bf16)sA0[v];
            pl[rr + 16 + lm]       = (__bf16)sA1[v];
            pl[512 + rr + lm]      = (__bf16)sB0[v];
            pl[512 + rr + 16 + lm] = (__bf16)sB1[v];
        }
        const __bf16* prowA = pl + lm * 32;
        const __bf16* prowB = prowA + 512;
        v16bf apA = frag16(prowA + lg * 16, prowA + lg * 16 + 8);
        v16bf apB = frag16(prowB + lg * 16, prowB + lg * 16 + 8);

        // ---- context accumulation: O += P @ V, V frags reused by A/B ----
        const __bf16* vp = vbase + k0 + lg * 16;
        {
            v16bf vf = fragB(vp + (size_t)(lm) * SEQ);
            OA0 = WMMA_BF16(apA, vf, OA0);  OB0 = WMMA_BF16(apB, vf, OB0);
        }
        {
            v16bf vf = fragB(vp + (size_t)(16 + lm) * SEQ);
            OA1 = WMMA_BF16(apA, vf, OA1);  OB1 = WMMA_BF16(apB, vf, OB1);
        }
        {
            v16bf vf = fragB(vp + (size_t)(32 + lm) * SEQ);
            OA2 = WMMA_BF16(apA, vf, OA2);  OB2 = WMMA_BF16(apB, vf, OB2);
        }
        {
            v16bf vf = fragB(vp + (size_t)(48 + lm) * SEQ);
            OA3 = WMMA_BF16(apA, vf, OA3);  OB3 = WMMA_BF16(apB, vf, OB3);
        }
    }

    // ---- normalize and write [B, S, H*64] fp32 --------------------------
#pragma unroll
    for (int v = 0; v < 8; ++v) {
        float invA = 1.0f / lA[v];
        float invB = 1.0f / lB[v];
        size_t rowA = ((size_t)b * SEQ + q0 + v + 8 * lg) * (NH * HD) + h * HD + lm;
        size_t rowB = rowA + (size_t)16 * (NH * HD);
        out[rowA +  0] = OA0[v] * invA;
        out[rowA + 16] = OA1[v] * invA;
        out[rowA + 32] = OA2[v] * invA;
        out[rowA + 48] = OA3[v] * invA;
        out[rowB +  0] = OB0[v] * invB;
        out[rowB + 16] = OB1[v] * invB;
        out[rowB + 32] = OB2[v] * invB;
        out[rowB + 48] = OB3[v] * invB;
    }
}

// ---------------------------------------------------------------------------
// Host launcher
// ---------------------------------------------------------------------------
extern "C" void kernel_launch(void* const* d_in, const int* in_sizes, int n_in,
                              void* d_out, int out_size, void* d_ws, size_t ws_size,
                              hipStream_t stream) {
    (void)in_sizes; (void)n_in; (void)out_size; (void)ws_size;
    const float* query      = (const float*)d_in[0];
    const float* key        = (const float*)d_in[1];
    const float* value      = (const float*)d_in[2];
    const float* key_mask   = (const float*)d_in[3];
    const float* Wq         = (const float*)d_in[4];
    const float* Wk         = (const float*)d_in[5];
    const float* Wv         = (const float*)d_in[6];
    const float* bias_table = (const float*)d_in[7];
    float* out = (float*)d_out;

    // workspace layout (bytes)
    char* ws = (char*)d_ws;
    const size_t WT_BYTES  = (size_t)3 * DM * DM * sizeof(__bf16);          //  6 MB
    const size_t XBF_BYTES = (size_t)3 * NB_ * SEQ * DM * sizeof(__bf16);   // 24 MB
    const size_t QKV_BYTES = (size_t)NB_ * NH * SEQ * HD * sizeof(__bf16);  //  8 MB each
    __bf16* wt    = (__bf16*)(ws);
    __bf16* xbf   = (__bf16*)(ws + WT_BYTES);
    __bf16* qws   = (__bf16*)(ws + WT_BYTES + XBF_BYTES);
    __bf16* kws   = (__bf16*)(ws + WT_BYTES + XBF_BYTES + QKV_BYTES);
    __bf16* vtws  = (__bf16*)(ws + WT_BYTES + XBF_BYTES + 2 * QKV_BYTES);
    float*  biasT = (float*) (ws + WT_BYTES + XBF_BYTES + 3 * QKV_BYTES);

    // 1) weights -> bf16 transposed
    t5_prep_weights<<<(3 * DM * DM) / 256, 256, 0, stream>>>(Wq, Wk, Wv, wt);
    // 2) activations -> bf16
    t5_prep_x<<<(3 * NB_ * SEQ * DM) / 256, 256, 0, stream>>>(query, key, value, xbf);
    // 3) relative-position bias table
    t5_prep_bias<<<(NH * 4095 + 255) / 256, 256, 0, stream>>>(bias_table, biasT);
    // 4) QKV projections: 3 * 256 row-tiles * 16 heads, 4 waves/block
    t5_proj_gemm<<<(3 * 256 * 16) / 4, 128, 0, stream>>>(xbf, wt, qws, kws, vtws);
    // 5) fused flash attention: 2*16*64 q-tiles (32 rows each), 4 waves/block
    t5_flash_attn<<<(NB_ * NH * 64) / 4, 128, 0, stream>>>(qws, kws, vtws,
                                                           key_mask, biasT, out);
}
